// RSR_2499670966386
// MI455X (gfx1250) — matrix operand order repaired and checked by reading, not compiled, not run on hardware
//
#include <hip/hip_runtime.h>
#include <hip/hip_bf16.h>
#include <math.h>

typedef __bf16 bf16_t;
typedef __attribute__((ext_vector_type(16))) __bf16 v16bf;
typedef __attribute__((ext_vector_type(8)))  __bf16 v8bf;
typedef __attribute__((ext_vector_type(8)))  float  v8f;
typedef __attribute__((ext_vector_type(4)))  float  f4;

#define T_STEPS 60
#define NEG_VAL  (-10000.0f)
#define LK_SLOPE (0.01f)

__device__ __forceinline__ v8f wmma_bf16(v16bf a, v16bf b, v8f c) {
  // v_wmma_f32_16x16x32_bf16  (neg_a, A, neg_b, B, c_mod, C, reuse_a, reuse_b)
  return __builtin_amdgcn_wmma_f32_16x16x32_bf16(false, a, false, b, (short)0, c, false, false);
}

__device__ __forceinline__ v16bf cat16(v8bf lo, v8bf hi) {
  return __builtin_shufflevector(lo, hi, 0,1,2,3,4,5,6,7,8,9,10,11,12,13,14,15);
}

// A-fragment (16x32 bf16) from a 16x64 bf16 LDS tile [m][k].
// lane<16 : halves 0..15 -> K=kt*32+0..15
// lane>=16: halves 0..15 -> K=kt*32+16..31
__device__ __forceinline__ v16bf load_afrag(const bf16_t* sH, int m, int kt, int half) {
  int ko = kt * 32 + half * 16;
  v8bf lo = *(const v8bf*)(sH + m * 64 + ko);
  v8bf hi = *(const v8bf*)(sH + m * 64 + ko + 8);
  return cat16(lo, hi);
}

// B-fragment from pre-swizzled image: img[lane][h] contiguous 16 halves per lane.
__device__ __forceinline__ v16bf load_bfrag(const bf16_t* img, int lane) {
  v8bf lo = *(const v8bf*)(img + lane * 16);
  v8bf hi = *(const v8bf*)(img + lane * 16 + 8);
  return cat16(lo, hi);
}

// Fast activations: raw v_rcp_f32 / v_exp_f32 (no IEEE divide on the
// sequential GRU critical path); use gfx1250 v_tanh_f32 if available.
__device__ __forceinline__ float fast_rcp(float x) { return __builtin_amdgcn_rcpf(x); }
__device__ __forceinline__ float sigmf(float x)    { return fast_rcp(1.0f + __expf(-x)); }
#if __has_builtin(__builtin_amdgcn_tanh_f32)
__device__ __forceinline__ float tanhfast(float x) { return __builtin_amdgcn_tanh_f32(x); }
#else
__device__ __forceinline__ float tanhfast(float x) {
  return 1.0f - 2.0f * fast_rcp(1.0f + __expf(2.0f * x));
}
#endif

// ---------------------------------------------------------------------------
// Kernel 1: fused 2-layer GRU over T=60 steps, batch tile = 16 rows/block.
// wave w computes gate-column tiles {w, w+4, w+8} => r/z/n for cols [16w,16w+16)
// ---------------------------------------------------------------------------
__global__ __launch_bounds__(128, 1)
void gru_kernel(const float* __restrict__ x,
                const float* __restrict__ Wih0, const float* __restrict__ Whh0,
                const float* __restrict__ bih0, const float* __restrict__ bhh0,
                const float* __restrict__ Wih1, const float* __restrict__ Whh1,
                const float* __restrict__ bih1, const float* __restrict__ bhh1,
                float* __restrict__ hfin)
{
  __shared__ __align__(16) bf16_t sW0[2][12][32][16];  // Whh0 B-frag image 24 KB
  __shared__ __align__(16) bf16_t sW1[2][12][32][16];  // Whh1 B-frag image 24 KB
  __shared__ __align__(16) bf16_t sX[6][60][16];       // x tile [d][t][m]  11.25 KB
  __shared__ __align__(16) bf16_t sH0[16][64];         // h0 staging (bf16) 2 KB
  __shared__ __align__(16) bf16_t sH1[16][64];         // h1 staging (bf16) 2 KB

  const int tid  = threadIdx.x;
  const int row0 = blockIdx.x * 16;

  // Build B-fragment images for Whh0 / Whh1 :  B[k][n] = W[n][k]
  for (int e = tid; e < 2 * 12288; e += 128) {
    int mat = e / 12288;
    int r   = e - mat * 12288;
    int kt  = r / 6144;  r -= kt * 6144;
    int nt  = r / 512;   r -= nt * 512;
    int ln  = r / 16;    int h = r - ln * 16;
    int n   = nt * 16 + (ln & 15);
    int k   = kt * 32 + (ln >> 4) * 16 + h;
    const float* W = mat ? Whh1 : Whh0;
    bf16_t (*dst)[12][32][16] = mat ? sW1 : sW0;
    dst[kt][nt][ln][h] = (bf16_t)W[n * 64 + k];
  }
  // x tile: sX[d][t][m] = x[row0+m][d*60+t]
  for (int e = tid; e < 6 * 60 * 16; e += 128) {
    int d = e / 960; int r = e - d * 960; int t = r / 16; int m = r - t * 16;
    sX[d][t][m] = (bf16_t)x[(size_t)(row0 + m) * 360 + d * 60 + t];
  }
  for (int e = tid; e < 16 * 64; e += 128) {
    ((bf16_t*)sH0)[e] = (bf16_t)0.0f;
    ((bf16_t*)sH1)[e] = (bf16_t)0.0f;
  }

  const int lane = tid & 31;
  const int w    = tid >> 5;     // wave 0..3
  const int c    = lane & 15;
  const int half = lane >> 4;
  const int mb   = half * 8;     // D-frag rows: vgpr k -> m = k + 8*half

  const int nR = w * 16 + c, nZ = nR + 64, nN = nR + 128;

  const float bi0R = bih0[nR], bi0Z = bih0[nZ], bi0N = bih0[nN];
  const float bh0R = bhh0[nR], bh0Z = bhh0[nZ], bh0N = bhh0[nN];
  const float bi1R = bih1[nR], bi1Z = bih1[nZ], bi1N = bih1[nN];
  const float bh1R = bhh1[nR], bh1Z = bhh1[nZ], bh1N = bhh1[nN];

  float wiR[6], wiZ[6], wiN[6];
  #pragma unroll
  for (int d = 0; d < 6; ++d) {
    wiR[d] = Wih0[nR * 6 + d];
    wiZ[d] = Wih0[nZ * 6 + d];
    wiN[d] = Wih0[nN * 6 + d];
  }

  // Wih1 B-fragments kept in registers: [gate][kt]
  v16bf w1f[3][2];
  #pragma unroll
  for (int g = 0; g < 3; ++g) {
    int n = (w + g * 4) * 16 + c;
    #pragma unroll
    for (int kt = 0; kt < 2; ++kt) {
      int k0 = kt * 32 + half * 16;
      #pragma unroll
      for (int h = 0; h < 16; ++h) w1f[g][kt][h] = (bf16_t)Wih1[n * 64 + k0 + h];
    }
  }

  float hold0[8], hold1[8];
  #pragma unroll
  for (int k = 0; k < 8; ++k) { hold0[k] = 0.0f; hold1[k] = 0.0f; }

  __syncthreads();

  const bf16_t* sH0f = &sH0[0][0];
  const bf16_t* sH1f = &sH1[0][0];

  for (int t = 0; t < T_STEPS; ++t) {
    v16bf a0k0 = load_afrag(sH0f, c, 0, half);
    v16bf a0k1 = load_afrag(sH0f, c, 1, half);
    v16bf a1k0 = load_afrag(sH1f, c, 0, half);
    v16bf a1k1 = load_afrag(sH1f, c, 1, half);
    __syncthreads();   // all staging reads done before rewrites

    // ---- layer 0: hg = h0 @ Whh0^T
    v8f hR = {}, hZ = {}, hN = {};
    hR = wmma_bf16(a0k0, load_bfrag(&sW0[0][w    ][0][0], lane), hR);
    hZ = wmma_bf16(a0k0, load_bfrag(&sW0[0][w + 4][0][0], lane), hZ);
    hN = wmma_bf16(a0k0, load_bfrag(&sW0[0][w + 8][0][0], lane), hN);
    hR = wmma_bf16(a0k1, load_bfrag(&sW0[1][w    ][0][0], lane), hR);
    hZ = wmma_bf16(a0k1, load_bfrag(&sW0[1][w + 4][0][0], lane), hZ);
    hN = wmma_bf16(a0k1, load_bfrag(&sW0[1][w + 8][0][0], lane), hN);

    // xg0 = x_t @ Wih0^T + bih0 : 6 x ds_load_b128 (8 contiguous m per lane)
    float xr[8], xz[8], xn[8];
    #pragma unroll
    for (int k = 0; k < 8; ++k) { xr[k] = bi0R; xz[k] = bi0Z; xn[k] = bi0N; }
    #pragma unroll
    for (int d = 0; d < 6; ++d) {
      v8bf xv8 = *(const v8bf*)(&sX[d][t][mb]);
      #pragma unroll
      for (int k = 0; k < 8; ++k) {
        float xv = (float)xv8[k];
        xr[k] += xv * wiR[d]; xz[k] += xv * wiZ[d]; xn[k] += xv * wiN[d];
      }
    }

    #pragma unroll
    for (int k = 0; k < 8; ++k) {
      int m = mb + k;
      float rg = sigmf(xr[k] + hR[k] + bh0R);
      float zg = sigmf(xz[k] + hZ[k] + bh0Z);
      float ng = tanhfast(xn[k] + rg * (hN[k] + bh0N));
      float hh = ng + zg * (hold0[k] - ng);
      hold0[k] = hh;
      sH0[m][nR] = (bf16_t)hh;
    }
    __syncthreads();   // new h0 staged

    // ---- layer 1: xg1 = h0_new @ Wih1^T ; hg1 = h1 @ Whh1^T
    v16bf b0 = load_afrag(sH0f, c, 0, half);
    v16bf b1 = load_afrag(sH0f, c, 1, half);

    v8f xR = {}, xZ = {}, xN = {};
    xR = wmma_bf16(b0, w1f[0][0], xR); xR = wmma_bf16(b1, w1f[0][1], xR);
    xZ = wmma_bf16(b0, w1f[1][0], xZ); xZ = wmma_bf16(b1, w1f[1][1], xZ);
    xN = wmma_bf16(b0, w1f[2][0], xN); xN = wmma_bf16(b1, w1f[2][1], xN);

    v8f gR = {}, gZ = {}, gN = {};
    gR = wmma_bf16(a1k0, load_bfrag(&sW1[0][w    ][0][0], lane), gR);
    gZ = wmma_bf16(a1k0, load_bfrag(&sW1[0][w + 4][0][0], lane), gZ);
    gN = wmma_bf16(a1k0, load_bfrag(&sW1[0][w + 8][0][0], lane), gN);
    gR = wmma_bf16(a1k1, load_bfrag(&sW1[1][w    ][0][0], lane), gR);
    gZ = wmma_bf16(a1k1, load_bfrag(&sW1[1][w + 4][0][0], lane), gZ);
    gN = wmma_bf16(a1k1, load_bfrag(&sW1[1][w + 8][0][0], lane), gN);

    #pragma unroll
    for (int k = 0; k < 8; ++k) {
      int m = mb + k;
      float rg = sigmf(xR[k] + bi1R + gR[k] + bh1R);
      float zg = sigmf(xZ[k] + bi1Z + gZ[k] + bh1Z);
      float ng = tanhfast(xN[k] + bi1N + rg * (gN[k] + bh1N));
      float hh = ng + zg * (hold1[k] - ng);
      hold1[k] = hh;
      sH1[m][nR] = (bf16_t)hh;
    }
    __syncthreads();   // new h1 staged for next step
  }

  #pragma unroll
  for (int k = 0; k < 8; ++k)
    hfin[(size_t)(row0 + mb + k) * 64 + nR] = hold1[k];
}

// ---------------------------------------------------------------------------
// Kernel 2: sa[i] = h[i]·Wv[0:64], sb[i] = h[i]·Wv[64:128]
// ---------------------------------------------------------------------------
__global__ void sasb_kernel(const float* __restrict__ hfin, const float* __restrict__ W,
                            float* __restrict__ sa, float* __restrict__ sb)
{
  int i = blockIdx.x * blockDim.x + threadIdx.x;
  if (i >= 1024) return;
  float a = 0.0f, b = 0.0f;
  #pragma unroll 8
  for (int k = 0; k < 64; ++k) {
    float h = hfin[(size_t)i * 64 + k];
    a += h * W[k];
    b += h * W[64 + k];
  }
  sa[i] = a; sb[i] = b;
}

// ---------------------------------------------------------------------------
// Kernel 3: single-pass over relation row (268 MB total stream, HBM-bound):
// sr + mask + leaky-relu + masked softmax + agg = valid@h + final FC.
// One block per row i.
// ---------------------------------------------------------------------------
__global__ __launch_bounds__(256)
void attn_kernel(const float* __restrict__ rel, const float* __restrict__ W,
                 const float* __restrict__ bsc,
                 const float* __restrict__ fc_w, const float* __restrict__ fc_b,
                 const float* __restrict__ hfin,
                 const float* __restrict__ sa, const float* __restrict__ sb,
                 float* __restrict__ out)
{
  __shared__ float tb[1024];
  __shared__ float mb_[1024];
  __shared__ float wr[64];
  __shared__ float red[256];
  __shared__ float aggp[4][64];

  const int i = blockIdx.x;
  const int tid = threadIdx.x;

  if (tid < 64) wr[tid] = W[128 + tid];
  __syncthreads();

  const float sai = sa[i];
  const float bb  = bsc[0];

  float lmax = -3.0e38f;
  for (int j = tid; j < 1024; j += 256) {
    const f4* rp = (const f4*)(rel + ((size_t)i * 1024 + j) * 64);
    float s = 0.0f, msum = 0.0f;
    #pragma unroll
    for (int q = 0; q < 16; ++q) {
      f4 v = __builtin_nontemporal_load(rp + q);   // streamed once: NT hint
      s    += v.x * wr[4*q] + v.y * wr[4*q+1] + v.z * wr[4*q+2] + v.w * wr[4*q+3];
      msum += v.x + v.y + v.z + v.w;
    }
    float wgt  = sai + sb[j] + s + bb;
    wgt        = (wgt >= 0.0f) ? wgt : LK_SLOPE * wgt;
    float mask = (msum != 0.0f) ? 1.0f : 0.0f;
    float tmp  = mask * wgt;
    float tt   = (tmp == 0.0f) ? NEG_VAL : tmp;
    tb[j]  = tt;
    mb_[j] = mask;
    lmax = fmaxf(lmax, tt);
  }
  red[tid] = lmax; __syncthreads();
  for (int s2 = 128; s2 >= 1; s2 >>= 1) {
    if (tid < s2) red[tid] = fmaxf(red[tid], red[tid + s2]);
    __syncthreads();
  }
  const float mx = red[0];
  __syncthreads();

  float lsum = 0.0f;
  for (int j = tid; j < 1024; j += 256) {
    float e = __expf(tb[j] - mx);
    lsum += e;                 // denominator over ALL j (masked ones underflow)
    tb[j] = e * mb_[j];        // numerator * mask
  }
  red[tid] = lsum; __syncthreads();
  for (int s2 = 128; s2 >= 1; s2 >>= 1) {
    if (tid < s2) red[tid] += red[tid + s2];
    __syncthreads();
  }
  const float invS = 1.0f / red[0];
  __syncthreads();

  // agg[col] = sum_j tb[j] * h[j][col]   (h is L2-resident, coalesced reads)
  const int col = tid & 63, seg = tid >> 6;
  float p = 0.0f;
  for (int j = seg * 256; j < seg * 256 + 256; ++j)
    p += tb[j] * hfin[(size_t)j * 64 + col];
  aggp[seg][col] = p;
  __syncthreads();

  float contrib = 0.0f;
  if (tid < 64) {
    contrib = fc_w[tid] * hfin[(size_t)i * 64 + tid];
  } else if (tid < 128) {
    int k = tid - 64;
    float agg = (aggp[0][k] + aggp[1][k] + aggp[2][k] + aggp[3][k]) * invS;
    contrib = fc_w[64 + k] * agg;
  }
  red[tid] = contrib; __syncthreads();
  for (int s2 = 128; s2 >= 1; s2 >>= 1) {
    if (tid < s2) red[tid] += red[tid + s2];
    __syncthreads();
  }
  if (tid == 0) out[i] = red[0] + fc_b[0];
}

// ---------------------------------------------------------------------------
extern "C" void kernel_launch(void* const* d_in, const int* in_sizes, int n_in,
                              void* d_out, int out_size, void* d_ws, size_t ws_size,
                              hipStream_t stream) {
  const float* x    = (const float*)d_in[0];
  const float* rel  = (const float*)d_in[1];
  const float* W    = (const float*)d_in[2];
  const float* b    = (const float*)d_in[3];
  const float* fc_w = (const float*)d_in[4];
  const float* fc_b = (const float*)d_in[5];
  const float* Wih0 = (const float*)d_in[6];
  const float* Whh0 = (const float*)d_in[7];
  const float* bih0 = (const float*)d_in[8];
  const float* bhh0 = (const float*)d_in[9];
  const float* Wih1 = (const float*)d_in[10];
  const float* Whh1 = (const float*)d_in[11];
  const float* bih1 = (const float*)d_in[12];
  const float* bhh1 = (const float*)d_in[13];

  float* ws   = (float*)d_ws;
  float* hfin = ws;                  // 1024*64 floats
  float* sa   = ws + 1024 * 64;      // 1024 floats
  float* sb   = sa + 1024;           // 1024 floats

  gru_kernel<<<64, 128, 0, stream>>>(x, Wih0, Whh0, bih0, bhh0,
                                     Wih1, Whh1, bih1, bhh1, hfin);
  sasb_kernel<<<4, 256, 0, stream>>>(hfin, W, sa, sb);
  attn_kernel<<<1024, 256, 0, stream>>>(rel, W, b, fc_w, fc_b, hfin, sa, sb,
                                        (float*)d_out);
}
